// MaskedSelfAttention_14748917694968
// MI455X (gfx1250) — matrix operand order repaired and checked by reading, not compiled, not run on hardware
//
#include <hip/hip_runtime.h>

#define B_ 4
#define S_ 4096
#define E_ 1024
#define D_ 64
// 1/sqrt(64) * log2(e): scores land pre-scaled for exp2-domain softmax
#define QSCALE (0.125f * 1.44269504088896340736f)

typedef __attribute__((ext_vector_type(16))) __bf16 v16bf;
typedef __attribute__((ext_vector_type(8)))  float  v8f;
typedef __attribute__((ext_vector_type(8)))  __bf16 bf8;
typedef __attribute__((ext_vector_type(8)))  unsigned short u16x8;
typedef __attribute__((ext_vector_type(4)))  float  f4;
typedef __attribute__((ext_vector_type(4)))  unsigned v4u;
typedef __attribute__((ext_vector_type(8)))  int v8i_;
typedef __attribute__((ext_vector_type(4)))  int v4i_;

union ABf  { v16bf v; struct { bf8 lo; bf8 hi; } s; };
union BF8U { u16x8 u; bf8 b; };

static __device__ __forceinline__ float fexp2(float v) {
#if __has_builtin(__builtin_amdgcn_exp2f)
  return __builtin_amdgcn_exp2f(v);   // v_exp_f32 (native base-2)
#else
  return exp2f(v);
#endif
}

static __device__ __forceinline__ unsigned short f2bf(float f) {
  unsigned u = __float_as_uint(f);
  u += 0x7FFFu + ((u >> 16) & 1u);   // round-to-nearest-even
  return (unsigned short)(u >> 16);
}

// pack 8 f32 -> 8 bf16 (native pk cvt when available)
static __device__ __forceinline__ bf8 pk8(f4 a, f4 b) {
#if __has_builtin(__builtin_amdgcn_cvt_pk_bf16_f32)
  typedef __attribute__((ext_vector_type(2))) __bf16 v2bf;
  union { v2bf p[4]; bf8 v; } u;
  u.p[0] = __builtin_amdgcn_cvt_pk_bf16_f32(a[0], a[1]);
  u.p[1] = __builtin_amdgcn_cvt_pk_bf16_f32(a[2], a[3]);
  u.p[2] = __builtin_amdgcn_cvt_pk_bf16_f32(b[0], b[1]);
  u.p[3] = __builtin_amdgcn_cvt_pk_bf16_f32(b[2], b[3]);
  return u.v;
#else
  BF8U r;
  #pragma unroll
  for (int i = 0; i < 4; ++i) { r.u[i] = f2bf(a[i]); r.u[i + 4] = f2bf(b[i]); }
  return r.b;
#endif
}

// A-fragment (16x32 bf16) from bf16 memory: two 16-byte segments per lane.
static __device__ __forceinline__ v16bf load_a(const unsigned short* s0,
                                               const unsigned short* s1) {
  ABf u;
  u.s.lo = *(const bf8*)s0;
  u.s.hi = *(const bf8*)s1;
  return u.v;
}

// A-fragment from f32 memory with in-register bf16 conversion.
static __device__ __forceinline__ v16bf load_a_f32(const float* p0, const float* p1) {
  f4 a0 = *(const f4*)p0, a1 = *(const f4*)(p0 + 4);
  f4 b0 = *(const f4*)p1, b1 = *(const f4*)(p1 + 4);
  ABf u;
  u.s.lo = pk8(a0, a1);
  u.s.hi = pk8(b0, b1);
  return u.v;
}

// Async global->LDS copy, 16B per lane, tracked by ASYNCcnt (projection).
static __device__ __forceinline__ void async_b128(void* lds, const void* gaddr) {
  asm volatile("global_load_async_to_lds_b128 %0, %1, off"
               :: "v"((unsigned)(unsigned long long)lds),
                  "v"((unsigned long long)gaddr)
               : "memory");
}
#define WAIT_ASYNC_0() asm volatile("s_wait_asynccnt 0x0" ::: "memory")
#define WAIT_ASYNC_6() asm volatile("s_wait_asynccnt 0x6" ::: "memory")

// Tensor Data Mover: one-shot 2D tile load Global->LDS, tracked by TENSORcnt.
// Descriptor bit layout per CDNA5 ISA sec 8.3-8.6 (2D, 2-byte elements).
static __device__ __forceinline__ void tdm_load_2d(
    unsigned lds_addr, const void* gaddr,
    unsigned tensor_d0, unsigned tensor_d1,
    unsigned tile_d0, unsigned tile_d1, unsigned stride0) {
  const unsigned long long ga = (unsigned long long)gaddr;
  const v4u g0 = { 1u,                                  // count=1, user mode
                   lds_addr,                            // lds_addr (bytes)
                   (unsigned)ga,                        // global_addr[31:0]
                   ((unsigned)(ga >> 32) & 0x01FFFFFFu) // global_addr[56:32]
                       | (2u << 30) };                  // type=2 ("image")
  const v8i_ g1 = { (int)(1u << 16),                    // data_size=1 (2 bytes)
                    (int)((tensor_d0 & 0xFFFFu) << 16), // tensor_dim0[15:0]
                    (int)((tensor_d0 >> 16) | ((tensor_d1 & 0xFFFFu) << 16)),
                    (int)((tensor_d1 >> 16) | (tile_d0 << 16)),
                    (int)tile_d1,                       // tile_dim1, tile_dim2=0
                    (int)stride0,                       // tensor_dim0_stride lo32
                    0, 0 };
  const v4i_ gz = { 0, 0, 0, 0 };
#if __clang_major__ >= 23
  const v8i_ gz8 = { 0, 0, 0, 0, 0, 0, 0, 0 };
  __builtin_amdgcn_tensor_load_to_lds(g0, g1, gz, gz, gz8, 0);
#else
  __builtin_amdgcn_tensor_load_to_lds(g0, g1, gz, gz, 0);
#endif
}

__global__ void cvt_f32_bf16(const float* __restrict__ src,
                             unsigned short* __restrict__ dst, int n) {
  int i = blockIdx.x * blockDim.x + threadIdx.x;
  if (i < n) dst[i] = f2bf(src[i]);
}

// ---------------------------------------------------------------------------
// Fused QKV projection. Block = 4 waves = 64 rows. W k-slabs (192 rows x 32 k)
// double-buffered in LDS via async copies; x read directly as f32 and
// converted in-register (co-executes with WMMA).
// ---------------------------------------------------------------------------
__global__ __launch_bounds__(128) void qkv_proj(
    const float* __restrict__ x,
    const unsigned short* __restrict__ wb,
    unsigned short* __restrict__ qb,
    unsigned short* __restrict__ kb,
    unsigned short* __restrict__ vt) {
  __shared__ __align__(128) unsigned short wbuf[2][192][32];  // 2 x 12KB

  const int tid  = threadIdx.x;
  const int lane = tid & 31;
  const int wave = tid >> 5;
  const int row0 = (blockIdx.x * 4 + wave) * 16;
  const int half = (lane >= 16) ? 1 : 0;
  const int koff = half ? 16 : 0;   // A-fragment K offset for this lane group
  const int boff = half ? 16 : 0;   // B-fragment K offset for this lane group
  const int nlo  = lane & 15;

  auto stage_w = [&](int nb, int k0) {
    #pragma unroll
    for (int j = 0; j < 6; ++j) {
      const int c = tid + j * 128;       // 0..767 chunks of 16B
      const int row = c >> 2, seg = c & 3;
      async_b128(&wbuf[nb][row][seg * 8],
                 wb + (size_t)row * E_ + k0 + seg * 8);
    }
  };

  v8f zero = {0.f, 0.f, 0.f, 0.f, 0.f, 0.f, 0.f, 0.f};
  v8f acc[12];
  #pragma unroll
  for (int j = 0; j < 12; ++j) acc[j] = zero;

  const float* arow = x + (size_t)(row0 + nlo) * E_;

  stage_w(0, 0);
  v16bf a_cur = load_a_f32(arow + koff, arow + 8 + koff);

  for (int it = 0; it < E_ / 32; ++it) {
    const int k0 = it * 32;
    if (it + 1 < E_ / 32) {
      stage_w((it + 1) & 1, k0 + 32);
      WAIT_ASYNC_6();                     // previous slab's copies retired
    } else {
      WAIT_ASYNC_0();
    }
    __syncthreads();

    // pipeline next A fragment (global f32 + cvt) over this slab's WMMAs
    v16bf a_next = a_cur;
    if (it + 1 < E_ / 32)
      a_next = load_a_f32(arow + k0 + 32 + koff, arow + k0 + 40 + koff);

    const unsigned short (*wl)[32] = wbuf[it & 1];
    v16bf bcur = *(const v16bf*)&wl[nlo][boff];
    #pragma unroll
    for (int j = 0; j < 12; ++j) {        // j/4 = {q,k,v}, j%4 = N-tile of 16
      v16bf bnext = bcur;
      if (j + 1 < 12) {
        const int jn = j + 1;
        bnext = *(const v16bf*)&wl[(jn >> 2) * 64 + (jn & 3) * 16 + nlo][boff];
      }
      acc[j] = __builtin_amdgcn_wmma_f32_16x16x32_bf16(
          false, a_cur, false, bcur, (short)0, acc[j], false, false);
      bcur = bnext;
    }
    a_cur = a_next;
    __syncthreads();                      // protect slab before overwrite
  }

  const int rbase = row0 + (half ? 8 : 0);
  const int bidx  = row0 >> 12;            // / S_
  const int s0i   = (row0 & (S_ - 1)) + (half ? 8 : 0);
  #pragma unroll
  for (int n = 0; n < 4; ++n) {
    const int col = n * 16 + nlo;
    #pragma unroll
    for (int r = 0; r < 8; ++r) {
      qb[(size_t)(rbase + r) * D_ + col] = f2bf(acc[n][r] * QSCALE);
      kb[(size_t)(rbase + r) * D_ + col] = f2bf(acc[4 + n][r]);
    }
    u16x8 pk;
    #pragma unroll
    for (int r = 0; r < 8; ++r) pk[r] = f2bf(acc[8 + n][r]);
    *(u16x8*)(vt + (size_t)(bidx * D_ + col) * S_ + s0i) = pk;
  }
}

// ---------------------------------------------------------------------------
// Flash attention. Block = 4 waves = 64 consecutive query rows of one batch.
// K (32x64) and V^T (64x32) tiles double-buffered in LDS, staged by the
// Tensor Data Mover (wave 0 issues, TENSORcnt + barrier synchronize).
// ---------------------------------------------------------------------------
__global__ __launch_bounds__(128) void attn_kernel(
    const unsigned short* __restrict__ qb,
    const unsigned short* __restrict__ kb,
    const unsigned short* __restrict__ vt,
    float* __restrict__ out) {
  __shared__ __align__(128) unsigned short kbuf[2][32][64];   // 2 x 4KB
  __shared__ __align__(128) unsigned short vbuf[2][64][32];   // 2 x 4KB
  __shared__ __align__(32)  unsigned short pbuf[4][16 * 32];  // P staging, 1KB/wave

  const int tid  = threadIdx.x;
  const int lane = tid & 31;
  const int wave = tid >> 5;
  const int bidx = blockIdx.x >> 6;              // 64 blocks per batch
  const int qb0  = (blockIdx.x & 63) * 64;       // block's first query row
  const int qs   = qb0 + wave * 16;              // wave's first query row
  const int half = (lane >= 16) ? 1 : 0;
  const int koff = half ? 16 : 0;
  const int boff = half ? 16 : 0;
  const int nlo  = lane & 15;
  const int rbase = qs + (half ? 8 : 0);

  const unsigned short* kbase = kb + (size_t)bidx * S_ * D_;
  const unsigned short* vbase = vt + (size_t)bidx * D_ * S_;

  // one TDM descriptor each for the K tile and the V^T tile
  auto stage_kv = [&](int nb, int t0) {
    tdm_load_2d((unsigned)(unsigned long long)&kbuf[nb][0][0],
                kbase + (size_t)t0 * D_,
                /*tensor_d0=*/D_, /*tensor_d1=*/S_ - t0,
                /*tile_d0=*/D_, /*tile_d1=*/32, /*stride0=*/D_);
    tdm_load_2d((unsigned)(unsigned long long)&vbuf[nb][0][0],
                vbase + t0,
                /*tensor_d0=*/S_ - t0, /*tensor_d1=*/D_,
                /*tile_d0=*/32, /*tile_d1=*/D_, /*stride0=*/S_);
  };

  const unsigned short* qrow = qb + (size_t)(bidx * S_ + qs + nlo) * D_;
  const v16bf aq0 = load_a(qrow + koff,      qrow + 8 + koff);
  const v16bf aq1 = load_a(qrow + 32 + koff, qrow + 40 + koff);

  const v8f zero = {0.f, 0.f, 0.f, 0.f, 0.f, 0.f, 0.f, 0.f};
  v8f o0 = zero, o1 = zero, o2 = zero, o3 = zero;
  float m[8], l[8];
  #pragma unroll
  for (int r = 0; r < 8; ++r) { m[r] = -__builtin_inff(); l[r] = 0.f; }

  unsigned short* pb = pbuf[wave];
  const int nIter = qb0 / 32 + 2;                // covers keys < qb0 + 64

  if (wave == 0) stage_kv(0, 0);
  for (int it = 0; it < nIter; ++it) {
    const int t0 = it * 32;
    if (wave == 0) {
      if (it + 1 < nIter) {
        stage_kv((it + 1) & 1, t0 + 32);
        __builtin_amdgcn_s_wait_tensorcnt(2);    // previous tile's 2 DMAs done
      } else {
        __builtin_amdgcn_s_wait_tensorcnt(0);
      }
    }
    __syncthreads();
    const int cb = it & 1;

    // preload all 4 K fragments (DS loads pipeline ahead of the WMMAs)
    const v16bf bk00 = *(const v16bf*)&kbuf[cb][nlo][boff];
    const v16bf bk01 = *(const v16bf*)&kbuf[cb][nlo][32 + boff];
    const v16bf bk10 = *(const v16bf*)&kbuf[cb][16 + nlo][boff];
    const v16bf bk11 = *(const v16bf*)&kbuf[cb][16 + nlo][32 + boff];

    v8f s0 = __builtin_amdgcn_wmma_f32_16x16x32_bf16(
        false, aq0, false, bk00, (short)0, zero, false, false);
    s0 = __builtin_amdgcn_wmma_f32_16x16x32_bf16(
        false, aq1, false, bk01, (short)0, s0, false, false);
    v8f s1 = __builtin_amdgcn_wmma_f32_16x16x32_bf16(
        false, aq0, false, bk10, (short)0, zero, false, false);
    s1 = __builtin_amdgcn_wmma_f32_16x16x32_bf16(
        false, aq1, false, bk11, (short)0, s1, false, false);

    // issue V fragment loads now; they complete behind the softmax VALU chain
    const v16bf bv0 = *(const v16bf*)&vbuf[cb][nlo][boff];
    const v16bf bv1 = *(const v16bf*)&vbuf[cb][16 + nlo][boff];
    const v16bf bv2 = *(const v16bf*)&vbuf[cb][32 + nlo][boff];
    const v16bf bv3 = *(const v16bf*)&vbuf[cb][48 + nlo][boff];

    if (t0 + 31 > qs) {                          // tile touches/passes diagonal
      const int key0 = t0 + nlo;
      #pragma unroll
      for (int r = 0; r < 8; ++r) {
        const int grow = rbase + r;
        s0[r] = (key0 > grow)      ? -__builtin_inff() : s0[r];
        s1[r] = (key0 + 16 > grow) ? -__builtin_inff() : s1[r];
      }
    }

    // online softmax in exp2 domain (scores pre-scaled by log2e/sqrt(d))
    #pragma unroll
    for (int r = 0; r < 8; ++r) {
      float t = fmaxf(s0[r], s1[r]);
      t = fmaxf(t, __shfl_xor(t, 8, 16));
      t = fmaxf(t, __shfl_xor(t, 4, 16));
      t = fmaxf(t, __shfl_xor(t, 2, 16));
      t = fmaxf(t, __shfl_xor(t, 1, 16));
      const float mn = fmaxf(m[r], t);
      const float corr = fexp2(m[r] - mn);
      m[r] = mn;
      const float p0 = fexp2(s0[r] - mn);
      const float p1 = fexp2(s1[r] - mn);
      float rs = p0 + p1;
      rs += __shfl_xor(rs, 8, 16);
      rs += __shfl_xor(rs, 4, 16);
      rs += __shfl_xor(rs, 2, 16);
      rs += __shfl_xor(rs, 1, 16);
      l[r] = l[r] * corr + rs;
      o0[r] *= corr; o1[r] *= corr; o2[r] *= corr; o3[r] *= corr;
      const int M = r + (half ? 8 : 0);
      pb[M * 32 + nlo]      = f2bf(p0);          // C-layout -> row-major P
      pb[M * 32 + nlo + 16] = f2bf(p1);
    }

    // re-read P as an A fragment (DS in-order within the wave)
    const v16bf ap = load_a(pb + nlo * 32 + koff, pb + nlo * 32 + 8 + koff);

    // O += P(16x32) x V(32x64): 4 N-tiles from the shared (preloaded) V^T tile
    o0 = __builtin_amdgcn_wmma_f32_16x16x32_bf16(
        false, ap, false, bv0, (short)0, o0, false, false);
    o1 = __builtin_amdgcn_wmma_f32_16x16x32_bf16(
        false, ap, false, bv1, (short)0, o1, false, false);
    o2 = __builtin_amdgcn_wmma_f32_16x16x32_bf16(
        false, ap, false, bv2, (short)0, o2, false, false);
    o3 = __builtin_amdgcn_wmma_f32_16x16x32_bf16(
        false, ap, false, bv3, (short)0, o3, false, false);

    __syncthreads();                             // protect tiles before overwrite
  }

  float* orow = out + (size_t)bidx * S_ * D_;
  #pragma unroll
  for (int r = 0; r < 8; ++r) {
    const float inv = 1.f / l[r];
    const size_t ro = (size_t)(rbase + r) * D_ + nlo;
    orow[ro]      = o0[r] * inv;
    orow[ro + 16] = o1[r] * inv;
    orow[ro + 32] = o2[r] * inv;
    orow[ro + 48] = o3[r] * inv;
  }
}

extern "C" void kernel_launch(void* const* d_in, const int* in_sizes, int n_in,
                              void* d_out, int out_size, void* d_ws, size_t ws_size,
                              hipStream_t stream) {
  const float* x  = (const float*)d_in[0];
  const float* Wq = (const float*)d_in[1];
  const float* Wk = (const float*)d_in[2];
  const float* Wv = (const float*)d_in[3];
  float* out = (float*)d_out;

  unsigned short* ws = (unsigned short*)d_ws;
  const size_t NW = (size_t)D_ * E_;        // 65,536 per weight
  const size_t NR = (size_t)B_ * S_ * D_;   // 1,048,576 per q/k/v
  unsigned short* wb = ws;                  // bf16 [Wq|Wk|Wv]
  unsigned short* qb = wb + 3 * NW;         // bf16 q (pre-scaled)
  unsigned short* kb = qb + NR;             // bf16 k
  unsigned short* vt = kb + NR;             // bf16 v transposed [B][64][S]

  cvt_f32_bf16<<<(int)((NW + 255) / 256), 256, 0, stream>>>(Wq, wb,          (int)NW);
  cvt_f32_bf16<<<(int)((NW + 255) / 256), 256, 0, stream>>>(Wk, wb + NW,     (int)NW);
  cvt_f32_bf16<<<(int)((NW + 255) / 256), 256, 0, stream>>>(Wv, wb + 2 * NW, (int)NW);

  qkv_proj<<<(B_ * S_) / 64, 128, 0, stream>>>(x, wb, qb, kb, vt);
  attn_kernel<<<(B_ * S_ / 16) / 4, 128, 0, stream>>>(qb, kb, vt, out);
}
